// SelfAttention_20478404067633
// MI455X (gfx1250) — matrix operand order repaired and checked by reading, not compiled, not run on hardware
//
#include <hip/hip_runtime.h>
#include <hip/hip_bf16.h>
#include <math.h>

// ---------------------------------------------------------------------------
// Types for CDNA5 WMMA (wave32): v_wmma_f32_16x16x32_bf16
// ---------------------------------------------------------------------------
typedef __bf16 bf16_t;
typedef __attribute__((ext_vector_type(16))) __bf16 v16bf;
typedef __attribute__((ext_vector_type(8)))  float  v8f;

#define T_SEQ   4096
#define D_MODEL 1024
#define N_HEADS 8
#define D_HEAD  128

// ---------------------------------------------------------------------------
// Fragment loaders implementing the documented CDNA5 operand layouts
// (CDNA5 ISA §7.12.2, 16-bit A 16x32 / B 32x16, wave32).
//
// A (16x32, row-major source): lane L (L<16) holds row M=L, K=0..7 then 16..23;
// lane L>=16 holds row M=L-16, K=8..15 then 24..31.  Two 16B loads per lane.
//
// B operand from a row-major [N x K] matrix (i.e. B^T): lane L holds
// column N=L&15, K = (L>=16 ? 8 : 0) + (0..7, 16..23), same order as A -> two 16B loads.
// ---------------------------------------------------------------------------
__device__ inline v16bf load_frag_a(const bf16_t* __restrict__ A, int lda,
                                    int k0, int lane) {
    const int m = lane & 15;
    const int o = (lane >> 4) << 3;                 // 0 or 8
    const bf16_t* p = A + (size_t)m * lda + k0 + o;
    union { v16bf v; uint4 q[2]; } f;
    f.q[0] = *reinterpret_cast<const uint4*>(p);        // K: o .. o+7
    f.q[1] = *reinterpret_cast<const uint4*>(p + 16);   // K: o+16 .. o+23
    return f.v;
}

__device__ inline v16bf load_frag_b(const bf16_t* __restrict__ Bt, int ldb,
                                    int k0, int lane) {
    const int n = lane & 15;
    const int o = (lane >> 4) << 3;                 // 0 or 8
    const bf16_t* p = Bt + (size_t)n * ldb + k0 + o;
    union { v16bf v; uint4 q[2]; } f;
    f.q[0] = *reinterpret_cast<const uint4*>(p);        // K: o .. o+7
    f.q[1] = *reinterpret_cast<const uint4*>(p + 16);   // K: o+16 .. o+23
    return f.v;
}

template <int MT, int NT>
__device__ inline void zero_acc(v8f (&acc)[MT][NT]) {
#pragma unroll
    for (int i = 0; i < MT; ++i)
#pragma unroll
        for (int j = 0; j < NT; ++j)
#pragma unroll
            for (int e = 0; e < 8; ++e) acc[i][j][e] = 0.0f;
}

// NT GEMM core: C[MT*16 x NT*16] += A[rows, K] * Bt[rows, K]^T, bf16 in, f32 acc
template <int MT, int NT>
__device__ inline void gemm_nt(const bf16_t* __restrict__ A, int lda,
                               const bf16_t* __restrict__ Bt, int ldb,
                               int K, v8f (&acc)[MT][NT], int lane) {
    for (int k0 = 0; k0 < K; k0 += 32) {
        v16bf a[MT];
        v16bf b[NT];
#pragma unroll
        for (int i = 0; i < MT; ++i)
            a[i] = load_frag_a(A + (size_t)(i * 16) * lda, lda, k0, lane);
#pragma unroll
        for (int j = 0; j < NT; ++j)
            b[j] = load_frag_b(Bt + (size_t)(j * 16) * ldb, ldb, k0, lane);
#pragma unroll
        for (int i = 0; i < MT; ++i)
#pragma unroll
            for (int j = 0; j < NT; ++j)
                acc[i][j] = __builtin_amdgcn_wmma_f32_16x16x32_bf16(
                    false, a[i], false, b[j], (short)0, acc[i][j], false, false);
    }
}

// C/D layout (ISA): lane L, VGPR r -> M = r + (L>=16 ? 8 : 0), N = L & 15.
#define TILE_COORDS()                              \
    const int tid  = threadIdx.x;                  \
    const int lane = tid & 31;                     \
    const int wave = tid >> 5;                     \
    const int wm   = wave >> 1;                    \
    const int wn   = wave & 1;                     \
    const int m0   = blockIdx.y * 64 + wm * 32;    \
    const int n0   = blockIdx.x * 128 + wn * 64;   \
    const int nloc = lane & 15;                    \
    const int mhalf = (lane >> 4) * 8;

// ---------------------------------------------------------------------------
// fp32 -> bf16 convert
// ---------------------------------------------------------------------------
__global__ void k_conv(const float* __restrict__ in, bf16_t* __restrict__ out, int n) {
    int i = blockIdx.x * blockDim.x + threadIdx.x;
    if (i < n) out[i] = (bf16_t)in[i];
}

// ---------------------------------------------------------------------------
// Sinusoidal absolute-position bias: bias[t,s], s even -> sin, odd -> cos,
// angle = t / FREQ^(2*(s/2)/d), FREQ=1e4, d=1024
// ---------------------------------------------------------------------------
__global__ void k_bias(float* __restrict__ bias) {
    size_t idx = (size_t)blockIdx.x * blockDim.x + threadIdx.x;
    if (idx >= (size_t)T_SEQ * T_SEQ) return;
    int t = (int)(idx >> 12);
    int s = (int)(idx & 4095);
    int i = s >> 1;
    // FREQ^(-2i/d) = exp(-(2i/d)*ln(1e4))
    float w = expf((-2.0f * (float)i / (float)D_MODEL) * 9.210340371976184f);
    float ang = (float)t * w;
    bias[idx] = (s & 1) ? cosf(ang) : sinf(ang);
}

// ---------------------------------------------------------------------------
// QKV projections: Q/K row-major bf16 [T, H*dh]; V stored transposed
// Vt[h*dh+e, t] so that att*V is also an NT GEMM.
// ---------------------------------------------------------------------------
__global__ __launch_bounds__(128) void k_qkv(
    const bf16_t* __restrict__ xb,
    const bf16_t* __restrict__ Wqb, const bf16_t* __restrict__ Wkb,
    const bf16_t* __restrict__ Wvb,
    bf16_t* __restrict__ Qb, bf16_t* __restrict__ Kb, bf16_t* __restrict__ Vt) {
    TILE_COORDS();
    const bf16_t* W = (blockIdx.z == 0) ? Wqb : (blockIdx.z == 1) ? Wkb : Wvb;
    v8f acc[2][4];
    zero_acc(acc);
    gemm_nt<2, 4>(xb + (size_t)m0 * D_MODEL, D_MODEL,
                  W + (size_t)n0 * D_MODEL, D_MODEL, D_MODEL, acc, lane);
    if (blockIdx.z < 2) {
        bf16_t* out = (blockIdx.z == 0) ? Qb : Kb;
#pragma unroll
        for (int i = 0; i < 2; ++i)
#pragma unroll
            for (int j = 0; j < 4; ++j)
#pragma unroll
                for (int r = 0; r < 8; ++r) {
                    int gm = m0 + i * 16 + mhalf + r;
                    int gn = n0 + j * 16 + nloc;
                    out[(size_t)gm * D_MODEL + gn] = (bf16_t)acc[i][j][r];
                }
    } else {
#pragma unroll
        for (int i = 0; i < 2; ++i)
#pragma unroll
            for (int j = 0; j < 4; ++j)
#pragma unroll
                for (int r = 0; r < 8; ++r) {
                    int gm = m0 + i * 16 + mhalf + r;   // t
                    int gn = n0 + j * 16 + nloc;        // feature
                    Vt[(size_t)gn * T_SEQ + gm] = (bf16_t)acc[i][j][r];
                }
    }
}

// ---------------------------------------------------------------------------
// Scores for one head: S[t,s] = Q_h[t,:] . K_h[s,:] + bias[t,s]   (fp32)
// ---------------------------------------------------------------------------
__global__ __launch_bounds__(128) void k_scores(
    const bf16_t* __restrict__ Qb, const bf16_t* __restrict__ Kb,
    const float* __restrict__ bias, float* __restrict__ S, int head) {
    TILE_COORDS();
    v8f acc[2][4];
    zero_acc(acc);
    gemm_nt<2, 4>(Qb + (size_t)m0 * D_MODEL + head * D_HEAD, D_MODEL,
                  Kb + (size_t)n0 * D_MODEL + head * D_HEAD, D_MODEL,
                  D_HEAD, acc, lane);
#pragma unroll
    for (int i = 0; i < 2; ++i)
#pragma unroll
        for (int j = 0; j < 4; ++j)
#pragma unroll
            for (int r = 0; r < 8; ++r) {
                size_t gm = (size_t)(m0 + i * 16 + mhalf + r);
                size_t gn = (size_t)(n0 + j * 16 + nloc);
                S[gm * T_SEQ + gn] = acc[i][j][r] + bias[gm * T_SEQ + gn];
            }
}

// ---------------------------------------------------------------------------
// Row softmax stats + unnormalized probs: P = bf16(exp(S - rowmax)),
// invsum[t] = 1/sum_s exp(S-max). One block per row.
// ---------------------------------------------------------------------------
__global__ __launch_bounds__(256) void k_softmax(
    const float* __restrict__ S, bf16_t* __restrict__ P,
    float* __restrict__ rowmax, float* __restrict__ invsum) {
    __shared__ float red[256];
    const int t = blockIdx.x;
    const float* row = S + (size_t)t * T_SEQ;
    float mx = -3.4e38f;
    for (int s = threadIdx.x; s < T_SEQ; s += 256) mx = fmaxf(mx, row[s]);
    red[threadIdx.x] = mx;
    __syncthreads();
    for (int off = 128; off > 0; off >>= 1) {
        if (threadIdx.x < off)
            red[threadIdx.x] = fmaxf(red[threadIdx.x], red[threadIdx.x + off]);
        __syncthreads();
    }
    mx = red[0];
    __syncthreads();
    float sum = 0.0f;
    bf16_t* prow = P + (size_t)t * T_SEQ;
    for (int s = threadIdx.x; s < T_SEQ; s += 256) {
        float e = __expf(row[s] - mx);
        prow[s] = (bf16_t)e;
        sum += e;
    }
    red[threadIdx.x] = sum;
    __syncthreads();
    for (int off = 128; off > 0; off >>= 1) {
        if (threadIdx.x < off) red[threadIdx.x] += red[threadIdx.x + off];
        __syncthreads();
    }
    if (threadIdx.x == 0) {
        rowmax[t] = mx;
        invsum[t] = 1.0f / red[0];
    }
}

// ---------------------------------------------------------------------------
// y_h = softmax(S) @ V_h : NT GEMM P[T,T] x Vt[dh,T]^T, scaled by invsum,
// written into the concatenated head buffer Yc[t, h*dh+e] (bf16).
// ---------------------------------------------------------------------------
__global__ __launch_bounds__(128) void k_av(
    const bf16_t* __restrict__ P, const bf16_t* __restrict__ Vt,
    const float* __restrict__ invsum, bf16_t* __restrict__ Yc, int head) {
    TILE_COORDS();  // gridDim.x == 1 -> n0 in [0,128)
    v8f acc[2][4];
    zero_acc(acc);
    gemm_nt<2, 4>(P + (size_t)m0 * T_SEQ, T_SEQ,
                  Vt + (size_t)(head * D_HEAD + n0) * T_SEQ, T_SEQ,
                  T_SEQ, acc, lane);
#pragma unroll
    for (int i = 0; i < 2; ++i)
#pragma unroll
        for (int j = 0; j < 4; ++j)
#pragma unroll
            for (int r = 0; r < 8; ++r) {
                int gm = m0 + i * 16 + mhalf + r;
                int gn = n0 + j * 16 + nloc;
                Yc[(size_t)gm * D_MODEL + head * D_HEAD + gn] =
                    (bf16_t)(acc[i][j][r] * invsum[gm]);
            }
}

// ---------------------------------------------------------------------------
// Last head's normalized attention, recomputed from fp32 scores for accuracy.
// ---------------------------------------------------------------------------
__global__ void k_att_out(const float* __restrict__ S,
                          const float* __restrict__ rowmax,
                          const float* __restrict__ invsum,
                          float* __restrict__ att) {
    size_t idx = (size_t)blockIdx.x * blockDim.x + threadIdx.x;
    if (idx >= (size_t)T_SEQ * T_SEQ) return;
    int t = (int)(idx >> 12);
    att[idx] = __expf(S[idx] - rowmax[t]) * invsum[t];
}

// ---------------------------------------------------------------------------
// Output projection: y = Yc @ Wout^T (fp32 out)
// ---------------------------------------------------------------------------
__global__ __launch_bounds__(128) void k_out(
    const bf16_t* __restrict__ Yc, const bf16_t* __restrict__ Wob,
    float* __restrict__ y) {
    TILE_COORDS();
    v8f acc[2][4];
    zero_acc(acc);
    gemm_nt<2, 4>(Yc + (size_t)m0 * D_MODEL, D_MODEL,
                  Wob + (size_t)n0 * D_MODEL, D_MODEL, D_MODEL, acc, lane);
#pragma unroll
    for (int i = 0; i < 2; ++i)
#pragma unroll
        for (int j = 0; j < 4; ++j)
#pragma unroll
            for (int r = 0; r < 8; ++r) {
                int gm = m0 + i * 16 + mhalf + r;
                int gn = n0 + j * 16 + nloc;
                y[(size_t)gm * D_MODEL + gn] = acc[i][j][r];
            }
}

// ---------------------------------------------------------------------------
extern "C" void kernel_launch(void* const* d_in, const int* in_sizes, int n_in,
                              void* d_out, int out_size, void* d_ws, size_t ws_size,
                              hipStream_t stream) {
    (void)in_sizes; (void)n_in; (void)out_size; (void)ws_size;
    const float* x    = (const float*)d_in[0];
    const float* Wq   = (const float*)d_in[1];
    const float* Wk   = (const float*)d_in[2];
    const float* Wv   = (const float*)d_in[3];
    const float* Wout = (const float*)d_in[4];

    char* wsp = (char*)d_ws;
    auto take = [&](size_t bytes) -> char* {
        char* p = wsp;
        wsp += (bytes + 255) & ~(size_t)255;
        return p;
    };
    bf16_t* xb   = (bf16_t*)take((size_t)T_SEQ * D_MODEL * 2);
    bf16_t* Wqb  = (bf16_t*)take((size_t)D_MODEL * D_MODEL * 2);
    bf16_t* Wkb  = (bf16_t*)take((size_t)D_MODEL * D_MODEL * 2);
    bf16_t* Wvb  = (bf16_t*)take((size_t)D_MODEL * D_MODEL * 2);
    bf16_t* Wob  = (bf16_t*)take((size_t)D_MODEL * D_MODEL * 2);
    bf16_t* Qb   = (bf16_t*)take((size_t)T_SEQ * D_MODEL * 2);
    bf16_t* Kb   = (bf16_t*)take((size_t)T_SEQ * D_MODEL * 2);
    bf16_t* Vt   = (bf16_t*)take((size_t)D_MODEL * T_SEQ * 2);
    float*  bias = (float*)take((size_t)T_SEQ * T_SEQ * 4);
    float*  S    = (float*)take((size_t)T_SEQ * T_SEQ * 4);
    bf16_t* P    = (bf16_t*)take((size_t)T_SEQ * T_SEQ * 2);
    float*  rmax = (float*)take((size_t)T_SEQ * 4);
    float*  isum = (float*)take((size_t)T_SEQ * 4);
    bf16_t* Yc   = (bf16_t*)take((size_t)T_SEQ * D_MODEL * 2);

    float* y_out   = (float*)d_out;
    float* att_out = (float*)d_out + (size_t)T_SEQ * D_MODEL;

    const int nX = T_SEQ * D_MODEL;
    const int nW = D_MODEL * D_MODEL;
    k_conv<<<(nX + 255) / 256, 256, 0, stream>>>(x, xb, nX);
    k_conv<<<(nW + 255) / 256, 256, 0, stream>>>(Wq, Wqb, nW);
    k_conv<<<(nW + 255) / 256, 256, 0, stream>>>(Wk, Wkb, nW);
    k_conv<<<(nW + 255) / 256, 256, 0, stream>>>(Wv, Wvb, nW);
    k_conv<<<(nW + 255) / 256, 256, 0, stream>>>(Wout, Wob, nW);

    k_bias<<<(T_SEQ * T_SEQ) / 256, 256, 0, stream>>>(bias);

    k_qkv<<<dim3(D_MODEL / 128, T_SEQ / 64, 3), 128, 0, stream>>>(
        xb, Wqb, Wkb, Wvb, Qb, Kb, Vt);

    for (int h = 0; h < N_HEADS; ++h) {
        k_scores<<<dim3(T_SEQ / 128, T_SEQ / 64), 128, 0, stream>>>(Qb, Kb, bias, S, h);
        k_softmax<<<T_SEQ, 256, 0, stream>>>(S, P, rmax, isum);
        k_av<<<dim3(1, T_SEQ / 64), 128, 0, stream>>>(P, Vt, isum, Yc, h);
        if (h == N_HEADS - 1)
            k_att_out<<<(T_SEQ * T_SEQ) / 256, 256, 0, stream>>>(S, rmax, isum, att_out);
    }

    k_out<<<dim3(D_MODEL / 128, T_SEQ / 64), 128, 0, stream>>>(Yc, Wob, y_out);
}
